// GAT_35820027248782
// MI455X (gfx1250) — matrix operand order repaired and checked by reading, not compiled, not run on hardware
//
#include <hip/hip_runtime.h>
#include <hip/hip_bf16.h>

#define N_NODES 100000
#define E_EDGES 1600000
#define ETOT    (E_EDGES + N_NODES)
#define IN_CH   128
#define HID     64
#define OUTC    40
#define SLOPE   0.2f

typedef __bf16 v16bf __attribute__((ext_vector_type(16)));
typedef __bf16 v2bf  __attribute__((ext_vector_type(2)));
typedef float  v8f   __attribute__((ext_vector_type(8)));
typedef int    v4i   __attribute__((ext_vector_type(4)));

// ---- fp32 pair -> packed bf16 dword (round-to-nearest-even) ----
static __device__ inline unsigned bf16rne(float f) {
  union { float f; unsigned u; } in; in.f = f;
  return (in.u + 0x7FFFu + ((in.u >> 16) & 1u)) >> 16;
}
static __device__ inline unsigned pk2bf(float lo, float hi) {
#if __has_builtin(__builtin_amdgcn_cvt_pk_bf16_f32)
  union { v2bf v; unsigned u; } cv;
  cv.v = __builtin_amdgcn_cvt_pk_bf16_f32(lo, hi);
  return cv.u;
#else
  return (bf16rne(lo) & 0xFFFFu) | (bf16rne(hi) << 16);
#endif
}

// ---- async global->LDS 16B copy (ASYNCcnt path), safe fallback ----
static __device__ inline void g2l_b128(const void* gsrc, void* lds_dst) {
#if __has_builtin(__builtin_amdgcn_global_load_async_to_lds_b128)
  __builtin_amdgcn_global_load_async_to_lds_b128(
      (__attribute__((address_space(1))) v4i*)gsrc,
      (__attribute__((address_space(3))) v4i*)lds_dst, 0, 0);
#else
  *(v4i*)lds_dst = *(const v4i*)gsrc;
#endif
}
static __device__ inline void wait_async_all() {
#if __has_builtin(__builtin_amdgcn_s_wait_asynccnt)
  __builtin_amdgcn_s_wait_asynccnt(0);
#elif __has_builtin(__builtin_amdgcn_global_load_async_to_lds_b128)
  asm volatile("s_wait_asynccnt 0" ::: "memory");
#endif
}

static __device__ inline void atomicMaxF(float* addr, float v) {
  if (v >= 0.0f) atomicMax((int*)addr, __float_as_int(v));
  else           atomicMin((unsigned int*)addr, __float_as_uint(v));
}

// fp32 -> packed bf16 (one dword = two consecutive elements)
__global__ void cvt_pk_bf16(const float* __restrict__ src,
                            unsigned* __restrict__ dst, int ndw) {
  int i = blockIdx.x * blockDim.x + threadIdx.x;
  if (i >= ndw) return;
  float2 f = ((const float2*)src)[i];
  dst[i] = pk2bf(f.x, f.y);
}

// ---------------------------------------------------------------------------
// h[M x COUT] = X[M x CIN] * W[COUT x CIN]^T with X,W pre-packed bf16 pairs.
// One block per 16-node row tile; the contiguous bf16 x-tile is staged into
// LDS via async global->LDS b128; zero per-iteration conversion VALU: A comes
// as two ds_load_b128, B as two global_load_b128 of packed dwords.
// ---------------------------------------------------------------------------
template <int CIN, int COUT>
__global__ void gat_gemm_wmma(const unsigned* __restrict__ xb,  // N x CIN/2 dw
                              const unsigned* __restrict__ wb,  // COUT x CIN/2 dw
                              float* __restrict__ h) {
  constexpr int CD  = CIN / 2;              // packed dwords per row
  constexpr int NCT = (COUT + 15) / 16;     // waves per block
  __shared__ __align__(16) unsigned tile[16 * CD];

  const int rowTile = blockIdx.x;
  const unsigned* gx = xb + (size_t)rowTile * (16 * CD);   // contiguous tile
  for (int idx = threadIdx.x * 4; idx < 16 * CD; idx += NCT * 32 * 4)
    g2l_b128(gx + idx, &tile[idx]);
  wait_async_all();
  __syncthreads();

  const int colTile = threadIdx.x >> 5;
  const int lane    = threadIdx.x & 31;
  const int hsel    = lane >> 4;            // upper/lower 16-lane half
  const int lm      = lane & 15;
  const int coln    = colTile * 16 + lm;    // out channel (may be OOB)
  const int colc    = (coln < COUT) ? coln : 0;
  const unsigned zm = (coln < COUT) ? 0xFFFFFFFFu : 0u;
  const unsigned* __restrict__ wrow = wb + (size_t)colc * CD;
  const unsigned* ar = &tile[lm * CD];

  union { v16bf v; unsigned u[8]; uint4 q[2]; } a, b;
  v8f c = {};
#pragma unroll
  for (int kd = 0; kd < CD; kd += 16) {     // kd = kt/2 (16 dw = 32 K-steps)
    // A 16x32 bf16: dwords [kd+hsel*4 .. +4) and [kd+8+hsel*4 .. +4)
    const uint4* ap = (const uint4*)(ar + kd + hsel * 4);
    a.q[0] = ap[0];
    a.q[1] = ap[2];
    // B 32x16 (= W^T): row coln, dwords [kd+hsel*4 .. +4) and [kd+8+hsel*4 .. +4)
    const uint4* wp = (const uint4*)(wrow + kd + hsel * 4);
    uint4 w0 = wp[0], w1 = wp[2];
    b.u[0] = w0.x & zm; b.u[1] = w0.y & zm;
    b.u[2] = w0.z & zm; b.u[3] = w0.w & zm;
    b.u[4] = w1.x & zm; b.u[5] = w1.y & zm;
    b.u[6] = w1.z & zm; b.u[7] = w1.w & zm;
    c = __builtin_amdgcn_wmma_f32_16x16x32_bf16(
            false, a.v, false, b.v, (short)0, c, false, false);
  }

  if (coln < COUT) {
#pragma unroll
    for (int r = 0; r < 8; ++r) {   // D layout: VGPR r -> M = r (+8 upper half)
      h[(size_t)(rowTile * 16 + r + hsel * 8) * COUT + coln] = c[r];
    }
  }
}

// Per-node attention scalars: a_dst[n] = att[0:C].h[n], a_src[n] = att[C:2C].h[n]
__global__ void gat_att_scalars(const float* __restrict__ h,
                                const float* __restrict__ att,
                                float* __restrict__ a_dst,
                                float* __restrict__ a_src, int Cout) {
  int n = blockIdx.x * blockDim.x + threadIdx.x;
  if (n >= N_NODES) return;
  const float* hr = h + (size_t)n * Cout;
  float sd = 0.0f, ss = 0.0f;
  for (int c = 0; c < Cout; ++c) {
    float v = hr[c];
    sd += att[c] * v;
    ss += att[Cout + c] * v;
  }
  a_dst[n] = sd;
  a_src[n] = ss;
}

__global__ void gat_fill(float* __restrict__ p, float v, int n) {
  int i = blockIdx.x * blockDim.x + threadIdx.x;
  if (i < n) p[i] = v;
}

// Segment max of leaky-relu(logit) keyed by destination node.
__global__ void gat_edge_max(const int* __restrict__ src,
                             const int* __restrict__ dst,
                             const float* __restrict__ a_dst,
                             const float* __restrict__ a_src,
                             float* __restrict__ m) {
  int e = blockIdx.x * blockDim.x + threadIdx.x;
  if (e >= ETOT) return;
  int s = (e < E_EDGES) ? src[e] : (e - E_EDGES);   // appended self loops
  int d = (e < E_EDGES) ? dst[e] : (e - E_EDGES);
  float l = a_dst[d] + a_src[s];
  l = (l > 0.0f) ? l : SLOPE * l;
  atomicMaxF(&m[d], l);
}

// One wave32 per edge: e = exp(lrelu(logit)-m[d]); s[d]+=e; out[d,:]+=e*h[src,:]
__global__ void gat_edge_accum(const int* __restrict__ src,
                               const int* __restrict__ dst,
                               const float* __restrict__ a_dst,
                               const float* __restrict__ a_src,
                               const float* __restrict__ m,
                               const float* __restrict__ h,
                               float* __restrict__ ssum,
                               float* __restrict__ out, int Cout) {
  int wid  = (blockIdx.x * blockDim.x + threadIdx.x) >> 5;
  int lane = threadIdx.x & 31;
  if (wid >= ETOT) return;
  int s = (wid < E_EDGES) ? src[wid] : (wid - E_EDGES);
  int d = (wid < E_EDGES) ? dst[wid] : (wid - E_EDGES);
  float l = a_dst[d] + a_src[s];
  l = (l > 0.0f) ? l : SLOPE * l;
  float e = __expf(l - m[d]);
  if (lane == 0) atomicAdd(&ssum[d], e);
  const float* __restrict__ hs = h + (size_t)s * Cout;
  float* __restrict__ od = out + (size_t)d * Cout;
  for (int c = lane; c < Cout; c += 32)
    atomicAdd(&od[c], e * hs[c]);
}

// dst_bf16_packed[n,c/2] = relu(out[n,c]/ssum[n]) : layer-0 epilogue fused with
// the bf16 packing of the layer-1 GEMM input. One thread per packed dword.
__global__ void gat_finalize_relu_bf16(const float* __restrict__ out,
                                       const float* __restrict__ ssum,
                                       unsigned* __restrict__ dst, int Cout) {
  int i = blockIdx.x * blockDim.x + threadIdx.x;   // packed dword index
  int ndw = N_NODES * (Cout / 2);
  if (i >= ndw) return;
  int n = i / (Cout / 2);
  float inv = 1.0f / ssum[n];
  float2 f = ((const float2*)out)[i];
  float v0 = f.x * inv; v0 = (v0 > 0.0f) ? v0 : 0.0f;
  float v1 = f.y * inv; v1 = (v1 > 0.0f) ? v1 : 0.0f;
  dst[i] = pk2bf(v0, v1);
}

// One wave32 per node: normalize by ssum, then log_softmax over OUTC classes.
__global__ void gat_logsoftmax(const float* __restrict__ agg,
                               const float* __restrict__ ssum,
                               float* __restrict__ y) {
  int n    = (blockIdx.x * blockDim.x + threadIdx.x) >> 5;
  int lane = threadIdx.x & 31;
  if (n >= N_NODES) return;
  float inv = 1.0f / ssum[n];
  const float* a = agg + (size_t)n * OUTC;
  float v0 = (lane      < OUTC) ? a[lane]      * inv : -__builtin_inff();
  float v1 = (lane + 32 < OUTC) ? a[lane + 32] * inv : -__builtin_inff();
  float mx = fmaxf(v0, v1);
#pragma unroll
  for (int o = 16; o; o >>= 1) mx = fmaxf(mx, __shfl_xor(mx, o, 32));
  float se = 0.0f;
  if (lane      < OUTC) se += __expf(v0 - mx);
  if (lane + 32 < OUTC) se += __expf(v1 - mx);
#pragma unroll
  for (int o = 16; o; o >>= 1) se += __shfl_xor(se, o, 32);
  float lse = __logf(se) + mx;
  if (lane      < OUTC) y[(size_t)n * OUTC + lane]      = v0 - lse;
  if (lane + 32 < OUTC) y[(size_t)n * OUTC + lane + 32] = v1 - lse;
}

extern "C" void kernel_launch(void* const* d_in, const int* in_sizes, int n_in,
                              void* d_out, int out_size, void* d_ws, size_t ws_size,
                              hipStream_t stream) {
  const float* x    = (const float*)d_in[0];
  const int*   ei   = (const int*)  d_in[1];
  const float* W0   = (const float*)d_in[2];
  const float* att0 = (const float*)d_in[3];
  const float* W1   = (const float*)d_in[4];
  const float* att1 = (const float*)d_in[5];
  float*       y    = (float*)d_out;

  const int* src = ei;             // edge_index[0]
  const int* dst = ei + E_EDGES;   // edge_index[1]

  // Workspace layout
  float*    bufA = (float*)d_ws;                         // h1 then h2 (N*HID f32)
  float*    bufB = bufA + (size_t)N_NODES * HID;         // out0 then out1
  unsigned* xb   = (unsigned*)(bufB + (size_t)N_NODES * HID); // N*(IN_CH/2) dw
  unsigned* g1b  = xb  + (size_t)N_NODES * (IN_CH / 2);  // N*(HID/2) dw
  unsigned* w0b  = g1b + (size_t)N_NODES * (HID / 2);    // HID*IN_CH/2 dw
  unsigned* w1b  = w0b + (size_t)HID * (IN_CH / 2);      // OUTC*HID/2 dw
  float*    adst = (float*)(w1b + (size_t)OUTC * (HID / 2));
  float*    asrc = adst + N_NODES;
  float*    mbuf = asrc + N_NODES;
  float*    sbuf = mbuf + N_NODES;

  const int rowTiles = N_NODES / 16;                     // 6250 exactly
  dim3 b256(256);

  // -------- bf16 packing passes (x, W0, W1) ------------------------------
  cvt_pk_bf16<<<dim3((N_NODES * (IN_CH / 2) + 255) / 256), b256, 0, stream>>>(
      x, xb, N_NODES * (IN_CH / 2));
  cvt_pk_bf16<<<dim3((HID * (IN_CH / 2) + 255) / 256), b256, 0, stream>>>(
      W0, w0b, HID * (IN_CH / 2));
  cvt_pk_bf16<<<dim3((OUTC * (HID / 2) + 255) / 256), b256, 0, stream>>>(
      W1, w1b, OUTC * (HID / 2));

  // -------- Layer 0: x(128) -> h1(64), attention aggregate, ReLU ----------
  gat_gemm_wmma<IN_CH, HID><<<dim3(rowTiles), dim3(32 * 4), 0, stream>>>(xb, w0b, bufA);
  gat_att_scalars<<<dim3((N_NODES + 255) / 256), b256, 0, stream>>>(bufA, att0, adst, asrc, HID);
  gat_fill<<<dim3((N_NODES + 255) / 256), b256, 0, stream>>>(mbuf, -__builtin_inff(), N_NODES);
  gat_fill<<<dim3((N_NODES + 255) / 256), b256, 0, stream>>>(sbuf, 0.0f, N_NODES);
  gat_fill<<<dim3((N_NODES * HID + 255) / 256), b256, 0, stream>>>(bufB, 0.0f, N_NODES * HID);
  gat_edge_max<<<dim3((ETOT + 255) / 256), b256, 0, stream>>>(src, dst, adst, asrc, mbuf);
  gat_edge_accum<<<dim3((ETOT + 7) / 8), b256, 0, stream>>>(src, dst, adst, asrc, mbuf,
                                                            bufA, sbuf, bufB, HID);
  gat_finalize_relu_bf16<<<dim3((N_NODES * (HID / 2) + 255) / 256), b256, 0, stream>>>(
      bufB, sbuf, g1b, HID);

  // -------- Layer 1: h1(64) -> h2(40), attention aggregate, log_softmax ---
  gat_gemm_wmma<HID, OUTC><<<dim3(rowTiles), dim3(32 * 3), 0, stream>>>(g1b, w1b, bufA);
  gat_att_scalars<<<dim3((N_NODES + 255) / 256), b256, 0, stream>>>(bufA, att1, adst, asrc, OUTC);
  gat_fill<<<dim3((N_NODES + 255) / 256), b256, 0, stream>>>(mbuf, -__builtin_inff(), N_NODES);
  gat_fill<<<dim3((N_NODES + 255) / 256), b256, 0, stream>>>(sbuf, 0.0f, N_NODES);
  gat_fill<<<dim3((N_NODES * OUTC + 255) / 256), b256, 0, stream>>>(bufB, 0.0f, N_NODES * OUTC);
  gat_edge_max<<<dim3((ETOT + 255) / 256), b256, 0, stream>>>(src, dst, adst, asrc, mbuf);
  gat_edge_accum<<<dim3((ETOT + 7) / 8), b256, 0, stream>>>(src, dst, adst, asrc, mbuf,
                                                            bufA, sbuf, bufB, OUTC);
  gat_logsoftmax<<<dim3((N_NODES + 7) / 8), b256, 0, stream>>>(bufB, sbuf, y);
}
